// Encoder_90967407330108
// MI455X (gfx1250) — compile-verified, matrix operand order checked by reading, not run on hardware
//
#include <hip/hip_runtime.h>

// ---------------------------------------------------------------------------
// Problem constants (match reference)
// ---------------------------------------------------------------------------
#define B_    32
#define N_    2048
#define C_    512
#define DW_   64
#define DBC_  128
#define KMAX_ 64

typedef __bf16 v16bf __attribute__((ext_vector_type(16)));
typedef __bf16 v8bf  __attribute__((ext_vector_type(8)));
typedef __bf16 v2bf  __attribute__((ext_vector_type(2)));
typedef float  v8f   __attribute__((ext_vector_type(8)));
typedef int    v4i   __attribute__((vector_size(16)));

#define AS1V __attribute__((address_space(1)))
#define AS3V __attribute__((address_space(3)))

// Async global->LDS (gfx1250, ASYNCcnt) when the toolchain exposes it.
#if defined(__has_builtin)
#  if __has_builtin(__builtin_amdgcn_global_load_async_to_lds_b128)
#    define USE_ASYNC_LDS 1
#  else
#    define USE_ASYNC_LDS 0
#  endif
#else
#  define USE_ASYNC_LDS 0
#endif

// ---------------------------------------------------------------------------
// WMMA fragment helpers (CDNA5 wave32 layouts, ISA 7.12.2)
//  A (16x32 bf16): lane = hi*16 + m ; elem e -> K = e<8 ? e+8*hi : 8+e+8*hi
//    => per lane: two contiguous 8-elem (16 B) runs at col0+8*hi, col0+16+8*hi
//  B (32x16 bf16): same K mapping; we PRE-SWIZZLE weights so each lane's 16
//    values are contiguous: frag f, lane l -> Bp[f*512 + l*16 + e]
//  C/D (16x16 f32): vgpr r, lane = hi*16 + n -> M = r + 8*hi
// ---------------------------------------------------------------------------
__device__ __host__ __forceinline__ int kmap(int e, int hi) {
    return (e < 8) ? (e + 8 * hi) : (8 + e + 8 * hi);
}

__device__ __forceinline__ v16bf load_a_frag(const __bf16* A, int lda, int row0, int col0) {
    const int lane = threadIdx.x & 31;
    const int m = lane & 15, hi = lane >> 4;
    const __bf16* p = A + (row0 + m) * lda + col0 + 8 * hi;
    v8bf a0 = *(const v8bf*)(p);
    v8bf a1 = *(const v8bf*)(p + 16);
    return __builtin_shufflevector(a0, a1, 0, 1, 2, 3, 4, 5, 6, 7,
                                   8, 9, 10, 11, 12, 13, 14, 15);
}

__device__ __forceinline__ v16bf load_b_packed(const __bf16* Bp, int frag) {
    const int lane = threadIdx.x & 31;
    return *(const v16bf*)(Bp + (frag * 32 + lane) * 16);
}

__device__ __forceinline__ v8f wmma_bf16(v16bf a, v16bf b, v8f c) {
    return __builtin_amdgcn_wmma_f32_16x16x32_bf16(false, a, false, b, (short)0, c,
                                                   false, false);
}

// ---------------------------------------------------------------------------
// Weight prep: f32 row-major (K x Nn) -> bf16 fragment-packed B layout.
//  frag f = kt*(Nn/16) + nt ; dst[f*512 + l*16 + e] = src[(kt*32+kmap(e,hi))*Nn + nt*16 + n]
// ---------------------------------------------------------------------------
__global__ void __launch_bounds__(256) swizzle_b(const float* __restrict__ src,
                                                 __bf16* __restrict__ dst,
                                                 int K, int Nn) {
    int idx = blockIdx.x * 256 + threadIdx.x;
    int total = K * Nn;
    if (idx >= total) return;
    int f = idx >> 9;
    int rem = idx & 511;
    int l = rem >> 4, e = rem & 15;
    int ntiles = Nn >> 4;
    int kt = f / ntiles, nt = f % ntiles;
    int n = l & 15, hi = l >> 4;
    int k = kt * 32 + kmap(e, hi);
    dst[idx] = (__bf16)src[k * Nn + nt * 16 + n];
}

// lp.W2 (32x1) -> fragment-packed 32x16 B tile with cols 1..15 zero.
__global__ void __launch_bounds__(256) pack_w2_sw(const float* __restrict__ w2,
                                                  __bf16* __restrict__ dst) {
    int idx = blockIdx.x * 256 + threadIdx.x;
    if (idx < 512) {
        int l = idx >> 4, e = idx & 15;
        int n = l & 15, hi = l >> 4;
        int k = kmap(e, hi);
        dst[idx] = (__bf16)((n == 0) ? w2[k] : 0.0f);
    }
}

// ---------------------------------------------------------------------------
// Front-end: wc = mlp2(data, wc), us = mlp2(data, us), fused over 128 out cols
//   grid = (B*N)/64 blocks, 256 threads (8 waves). Layer0 K=512 chunked by 64.
//   Packed weights: wcW0p/usW0p: 16 kt x 4 nt frags (chunk kc = 4096 contiguous
//   elements at kc*4096); wcW1p/usW1p: 2 kt x 4 nt frags (4096 elements).
// ---------------------------------------------------------------------------
__global__ void __launch_bounds__(256)
frontend_mlp2(const float* __restrict__ data,
              const __bf16* __restrict__ wcW0p, const __bf16* __restrict__ wcW1p,
              const __bf16* __restrict__ usW0p, const __bf16* __restrict__ usW1p,
              const float* __restrict__ wc_b0, const float* __restrict__ wc_b1,
              const float* __restrict__ wc_a0,
              const float* __restrict__ us_b0, const float* __restrict__ us_b1,
              const float* __restrict__ us_a0,
              float* __restrict__ wc_out, float* __restrict__ us_out)
{
    extern __shared__ char smem[];
    __bf16* sX  = (__bf16*)smem;        // 64 x 72 (A, row-major)
    __bf16* sWp = sX + 64 * 72;         // 8192 packed-frag weights (wc | us)
    __bf16* sH  = sWp + 8192;           // 64 x 136 hidden (A for layer1)
    const int LDX = 72, LDH = 136;

    const int tid  = threadIdx.x;
    const int wave = tid >> 5;
    const int lane = tid & 31;
    const long row0 = (long)blockIdx.x * 64;   // row over B*N

    const int j   = wave >> 2;          // 0 = wc, 1 = us
    const int ntL = wave & 3;           // column tile within the selected mlp

    v8f acc[4];
#pragma unroll
    for (int m = 0; m < 4; ++m) acc[m] = {};

    for (int kc = 0; kc < 8; ++kc) {
        if (kc < 7)
            __builtin_prefetch(&data[(row0 + (tid >> 2)) * C_ + (kc + 1) * 64], 0, 1);
        // A chunk: 64 rows x 64 cols f32 -> bf16 pairs
        for (int idx = tid; idx < 64 * 32; idx += 256) {
            int r = idx >> 5, cp = idx & 31;
            float2 d = ((const float2*)(data + (row0 + r) * C_ + kc * 64))[cp];
            v2bf pb; pb[0] = (__bf16)d.x; pb[1] = (__bf16)d.y;
            *(v2bf*)(sX + r * LDX + cp * 2) = pb;
        }
        // packed weight chunk (contiguous): wc -> sWp[0:4096), us -> sWp[4096:8192)
        {
            const uint4* gw = (const uint4*)(wcW0p + kc * 4096);
            const uint4* gu = (const uint4*)(usW0p + kc * 4096);
            uint4* lw = (uint4*)sWp;
            for (int q = tid; q < 512; q += 256) {
                lw[q]       = gw[q];
                lw[512 + q] = gu[q];
            }
        }
        __syncthreads();
#pragma unroll
        for (int ks = 0; ks < 2; ++ks) {
            v16bf bf = load_b_packed(sWp + j * 4096, ks * 4 + ntL);
#pragma unroll
            for (int m = 0; m < 4; ++m) {
                v16bf af = load_a_frag(sX, LDX, m * 16, ks * 32);
                acc[m] = wmma_bf16(af, bf, acc[m]);
            }
        }
        __syncthreads();
    }

    // bias + PReLU -> hidden (bf16) in LDS
    {
        const int n = lane & 15, hi = lane >> 4;
        const int col = wave * 16 + n;
        const float b0 = (col < 64) ? wc_b0[col] : us_b0[col - 64];
        const float a0 = (col < 64) ? wc_a0[0] : us_a0[0];
#pragma unroll
        for (int m = 0; m < 4; ++m)
#pragma unroll
            for (int r = 0; r < 8; ++r) {
                float v = acc[m][r] + b0;
                v = (v >= 0.0f) ? v : a0 * v;
                sH[(m * 16 + r + 8 * hi) * LDH + col] = (__bf16)v;
            }
    }
    // stage packed layer-1 weights
    {
        const uint4* gw = (const uint4*)wcW1p;
        const uint4* gu = (const uint4*)usW1p;
        uint4* lw = (uint4*)sWp;
        for (int q = tid; q < 512; q += 256) {
            lw[q]       = gw[q];
            lw[512 + q] = gu[q];
        }
    }
    __syncthreads();

    // layer 1: out(64x128) — wave<4: wc from H[:,0:64]; wave>=4: us from H[:,64:128]
    {
        v8f acc2[4];
#pragma unroll
        for (int m = 0; m < 4; ++m) acc2[m] = {};
#pragma unroll
        for (int ks = 0; ks < 2; ++ks) {
            v16bf bf = load_b_packed(sWp + j * 4096, ks * 4 + ntL);
#pragma unroll
            for (int m = 0; m < 4; ++m) {
                v16bf af = load_a_frag(sH, LDH, m * 16, j * 64 + ks * 32);
                acc2[m] = wmma_bf16(af, bf, acc2[m]);
            }
        }
        const int n = lane & 15, hi = lane >> 4;
        const int col = wave * 16 + n;
#pragma unroll
        for (int m = 0; m < 4; ++m)
#pragma unroll
            for (int r = 0; r < 8; ++r) {
                long gr = row0 + m * 16 + r + 8 * hi;
                if (col < 64) wc_out[gr * DW_ + col]        = acc2[m][r] + wc_b1[col];
                else          us_out[gr * DW_ + (col - 64)] = acc2[m][r] + us_b1[col - 64];
            }
    }
}

// ---------------------------------------------------------------------------
// In-place suffix sum: usbuf[b,n,:] <- sum_{m>n} us[b,m,:]
// ---------------------------------------------------------------------------
__global__ void __launch_bounds__(64) suffix_kernel(float* __restrict__ usb) {
    const int b = blockIdx.x, d = threadIdx.x;
    float carry = 0.0f;
    for (int n = N_ - 1; n >= 0; --n) {
        size_t idx = ((size_t)b * N_ + n) * DW_ + d;
        float v = usb[idx];
        usb[idx] = carry;
        carry += v;
    }
}

__global__ void cummax_kernel(const int* __restrict__ labels, int* __restrict__ cm) {
    if (blockIdx.x == 0 && threadIdx.x == 0) {
        int m = -1;
        for (int i = 0; i < N_; ++i) {
            int v = labels[i];
            m = (v > m) ? v : m;
            cm[i] = m;
        }
    }
}

// WC <- 0, WC[:,0,:] = wc[:,0,:], out <- 0
__global__ void __launch_bounds__(256)
init_state(float* __restrict__ WC, const float* __restrict__ wc, float* __restrict__ out) {
    long idx = (long)blockIdx.x * 256 + threadIdx.x;
    if (idx < (long)B_ * KMAX_ * DW_) {
        long b = idx / (KMAX_ * DW_);
        long rem = idx % (KMAX_ * DW_);
        long k = rem / DW_, d = rem % DW_;
        WC[idx] = (k == 0) ? wc[(b * N_) * DW_ + d] : 0.0f;
    }
    if (idx == 0) out[0] = 0.0f;
}

// ---------------------------------------------------------------------------
// Scan step (fused), one workgroup per batch, 8 waves.
// Packed weight block Wp (32256 bf16, contiguous):
//   [0,8192)      bcW0  (2 kt x 8 nt)
//   [8192,24576)  bcW1  (4 kt x 8 nt)
//   [24576,30720) lpW0  (6 kt x 2 nt)
//   [30720,31744) lpW1  (1 kt x 2 nt)
//   [31744,32256) lpW2  (1 frag, col0 = W2)
// ---------------------------------------------------------------------------
__global__ void __launch_bounds__(256)
step_kernel(int t,
            const float* __restrict__ wc, const float* __restrict__ US,
            const int* __restrict__ labels, const int* __restrict__ cm,
            float* __restrict__ WC,
            const __bf16* __restrict__ Wp,
            const float* __restrict__ bc_b0, const float* __restrict__ bc_b1,
            const float* __restrict__ bc_a0,
            const float* __restrict__ lp_b0, const float* __restrict__ lp_b1,
            const float* __restrict__ lp_a0, const float* __restrict__ lp_a1,
            float* __restrict__ out)
{
    const int b    = blockIdx.x;
    const int tid  = threadIdx.x;
    const int wave = tid >> 5;
    const int lane = tid & 31;
    const int i    = t + 2;
    const int lab_prev = labels[i - 1];
    const int lab_i    = labels[i];
    const int K        = cm[i - 1] + 1;    // K_arr[t]

    extern __shared__ char smem[];
    __bf16* sWp = (__bf16*)smem;           // 32256 packed weights
    __bf16* sA  = sWp + 32256;             // union: bc A 144x72 / lp A 80x200 (16000)
    __bf16* sHx = sA + 16000;              // union: bc H 144x136 / lp H 80x40 (19584)
    float*  sO  = (float*)(sHx + 19584);   // 144 x 132
    float*  sS  = sO + 144 * 132;          // 128
    float*  sWci= sS + 128;                // 64 (wc_i row, later US_i row)
    float*  sLg = sWci + 64;               // 80 logits

    const __bf16* sW0p = sWp;
    const __bf16* sW1p = sWp + 8192;
    const __bf16* sL0p = sWp + 24576;
    const __bf16* sL1p = sWp + 30720;
    const __bf16* sL2p = sWp + 31744;

    // 1) scatter-add WC[b, lab_prev] += wc[b, i-1]; stage wc_i
    if (tid < DW_) {
        WC[((long)b * KMAX_ + lab_prev) * DW_ + tid] += wc[((long)b * N_ + (i - 1)) * DW_ + tid];
        sWci[tid] = wc[((long)b * N_ + i) * DW_ + tid];
    }
    // 2) stage packed weight block: 32256 bf16 = 4032 x 16B
#if USE_ASYNC_LDS
    {
        const AS1V char* g1 = (const AS1V char*)(const char*)Wp;
        AS3V char*       l1 = (AS3V char*)(char*)sWp;
        AS1V v4i* g = (AS1V v4i*)g1;   // exact param type: v4i addrspace(1)*
        AS3V v4i* l = (AS3V v4i*)l1;   // v4i addrspace(3)*
        for (int q = tid; q < 4032; q += 256)
            __builtin_amdgcn_global_load_async_to_lds_b128(g + q, l + q, 0, 0);
        asm volatile("s_wait_asynccnt 0x0" ::: "memory");
    }
#else
    {
        const uint4* g = (const uint4*)Wp;
        uint4* l = (uint4*)sWp;
        for (int q = tid; q < 4032; q += 256) l[q] = g[q];
    }
#endif
    __syncthreads();

    // 3) build bc A matrix (144 x 64, ld 72), bf16 pairs
    for (int idx = tid; idx < 144 * 32; idx += 256) {
        int r = idx >> 5, dp = idx & 31;
        float2 v;
        if (r < 64)
            v = ((const float2*)(WC + ((long)b * KMAX_ + r) * DW_))[dp];
        else if (r < 128) {
            float2 w = ((const float2*)(WC + ((long)b * KMAX_ + (r - 64)) * DW_))[dp];
            float2 c = ((const float2*)sWci)[dp];
            v = make_float2(w.x + c.x, w.y + c.y);
        } else if (r == 128)
            v = ((const float2*)sWci)[dp];
        else
            v = make_float2(0.0f, 0.0f);
        v2bf pb; pb[0] = (__bf16)v.x; pb[1] = (__bf16)v.y;
        *(v2bf*)(sA + r * 72 + dp * 2) = pb;
    }
    __syncthreads();

    // 4) bc layer 0: (144x64)@(64x128); 9x8 tiles, wave owns column tile nt==wave
    {
        v8f acc[9];
#pragma unroll
        for (int m = 0; m < 9; ++m) acc[m] = {};
#pragma unroll
        for (int ks = 0; ks < 2; ++ks) {
            v16bf bf = load_b_packed(sW0p, ks * 8 + wave);
#pragma unroll
            for (int m = 0; m < 9; ++m) {
                v16bf af = load_a_frag(sA, 72, m * 16, ks * 32);
                acc[m] = wmma_bf16(af, bf, acc[m]);
            }
        }
        const int n = lane & 15, hi = lane >> 4;
        const int col = wave * 16 + n;
        const float b0 = bc_b0[col], a0 = bc_a0[0];
#pragma unroll
        for (int m = 0; m < 9; ++m)
#pragma unroll
            for (int r = 0; r < 8; ++r) {
                float v = acc[m][r] + b0;
                v = (v >= 0.0f) ? v : a0 * v;
                sHx[(m * 16 + r + 8 * hi) * 136 + col] = (__bf16)v;
            }
    }
    __syncthreads();

    // 5) bc layer 1: (144x128)@(128x128) -> sO (f32)
    {
        v8f acc[9];
#pragma unroll
        for (int m = 0; m < 9; ++m) acc[m] = {};
#pragma unroll
        for (int ks = 0; ks < 4; ++ks) {
            v16bf bf = load_b_packed(sW1p, ks * 8 + wave);
#pragma unroll
            for (int m = 0; m < 9; ++m) {
                v16bf af = load_a_frag(sHx, 136, m * 16, ks * 32);
                acc[m] = wmma_bf16(af, bf, acc[m]);
            }
        }
        const int n = lane & 15, hi = lane >> 4;
        const int col = wave * 16 + n;
        const float b1 = bc_b1[col];
#pragma unroll
        for (int m = 0; m < 9; ++m)
#pragma unroll
            for (int r = 0; r < 8; ++r)
                sO[(m * 16 + r + 8 * hi) * 132 + col] = acc[m][r] + b1;
    }
    __syncthreads();

    // 6) S[d] = sum_{k<K} h[k][d]; stage US_i row into sWci
    if (tid < DBC_) {
        float s = 0.0f;
        for (int k = 0; k < K; ++k) s += sO[k * 132 + tid];
        sS[tid] = s;
    } else if (tid < DBC_ + DW_) {
        sWci[tid - DBC_] = US[((long)b * N_ + i) * DW_ + (tid - DBC_)];
    }
    __syncthreads();

    // 7) build lp A (80 x 192, ld 200): rows 0..63 = [US_i | BC_k], row 64 = [US_i | BC_new]
    for (int idx = tid; idx < 80 * 96; idx += 256) {
        int r = idx / 96, cp = idx % 96;
        int c = cp * 2;
        float2 v;
        if (r < 64) {
            if (c < 64) v = ((const float2*)sWci)[cp];
            else {
                int q = (c - 64) >> 1;
                float2 s  = ((const float2*)sS)[q];
                float2 hd = ((const float2*)(sO + (64 + r) * 132))[q];
                float2 h  = ((const float2*)(sO + r * 132))[q];
                v = make_float2(s.x + hd.x - h.x, s.y + hd.y - h.y);
            }
        } else if (r == 64) {
            if (c < 64) v = ((const float2*)sWci)[cp];
            else {
                int q = (c - 64) >> 1;
                float2 s  = ((const float2*)sS)[q];
                float2 o  = ((const float2*)(sO + 128 * 132))[q];
                v = make_float2(s.x + o.x, s.y + o.y);
            }
        } else
            v = make_float2(0.0f, 0.0f);
        v2bf pb; pb[0] = (__bf16)v.x; pb[1] = (__bf16)v.y;
        *(v2bf*)(sA + r * 200 + c) = pb;
    }
    __syncthreads();

    // 8) lp layer 0: (80x192)@(192x32); 10 tiles over 8 waves
    for (int tt = wave; tt < 10; tt += 8) {
        const int mt = tt % 5, nt = tt / 5;
        v8f acc = {};
#pragma unroll
        for (int ks = 0; ks < 6; ++ks) {
            v16bf af = load_a_frag(sA, 200, mt * 16, ks * 32);
            v16bf bf = load_b_packed(sL0p, ks * 2 + nt);
            acc = wmma_bf16(af, bf, acc);
        }
        const int n = lane & 15, hi = lane >> 4;
        const int col = nt * 16 + n;
        const float b0 = lp_b0[col], a0 = lp_a0[0];
#pragma unroll
        for (int r = 0; r < 8; ++r) {
            float v = acc[r] + b0;
            v = (v >= 0.0f) ? v : a0 * v;
            sHx[(mt * 16 + r + 8 * hi) * 40 + col] = (__bf16)v;
        }
    }
    __syncthreads();

    // 9) lp layer 1: (80x32)@(32x32) in place (compute, barrier, write)
    {
        v8f zc = {};
        v8f acc0 = zc, acc1 = zc;
        {
            const int mt = wave % 5, nt = wave / 5;
            v16bf af = load_a_frag(sHx, 40, mt * 16, 0);
            v16bf bf = load_b_packed(sL1p, nt);
            acc0 = wmma_bf16(af, bf, zc);
        }
        const bool two = (wave < 2);
        if (two) {
            const int tt = wave + 8, mt = tt % 5, nt = tt / 5;
            v16bf af = load_a_frag(sHx, 40, mt * 16, 0);
            v16bf bf = load_b_packed(sL1p, nt);
            acc1 = wmma_bf16(af, bf, zc);
        }
        __syncthreads();   // all reads of sHx complete before overwrite
        const int n = lane & 15, hi = lane >> 4;
        {
            const int mt = wave % 5, nt = wave / 5;
            const int col = nt * 16 + n;
            const float b1 = lp_b1[col], a1 = lp_a1[0];
#pragma unroll
            for (int r = 0; r < 8; ++r) {
                float v = acc0[r] + b1;
                v = (v >= 0.0f) ? v : a1 * v;
                sHx[(mt * 16 + r + 8 * hi) * 40 + col] = (__bf16)v;
            }
        }
        if (two) {
            const int tt = wave + 8, mt = tt % 5, nt = tt / 5;
            const int col = nt * 16 + n;
            const float b1 = lp_b1[col], a1 = lp_a1[0];
#pragma unroll
            for (int r = 0; r < 8; ++r) {
                float v = acc1[r] + b1;
                v = (v >= 0.0f) ? v : a1 * v;
                sHx[(mt * 16 + r + 8 * hi) * 40 + col] = (__bf16)v;
            }
        }
    }
    __syncthreads();

    // 10) lp layer 2: (80x32)@(32x16 padded); only column 0 is the logit
    if (wave < 5) {
        const int mt = wave;
        v16bf af = load_a_frag(sHx, 40, mt * 16, 0);
        v16bf bf = load_b_packed(sL2p, 0);
        v8f zc = {};
        v8f acc = wmma_bf16(af, bf, zc);
        const int n = lane & 15, hi = lane >> 4;
        if (n == 0) {
#pragma unroll
            for (int r = 0; r < 8; ++r)
                sLg[mt * 16 + r + 8 * hi] = acc[r];
        }
    }
    __syncthreads();

    // 11) masked logsumexp + NLL; mean over B folded into the atomic
    if (tid == 0) {
        float mx = -1e30f;
        for (int j = 0; j <= KMAX_; ++j) {
            float v = (j <= K) ? sLg[j] : -1e30f;
            sLg[j] = v;
            mx = (v > mx) ? v : mx;
        }
        float se = 0.0f;
        for (int j = 0; j <= KMAX_; ++j) se += expf(sLg[j] - mx);
        float lse = mx + logf(se);
        float nll = -(sLg[lab_i] - lse);
        atomicAdd(out, nll * (1.0f / (float)B_));
    }
}

// ---------------------------------------------------------------------------
// Host launcher.  Input order = jax pytree flatten (dict keys sorted):
//  0 data, 1 labels, then params: bc{W0,W1,a0,b0,b1}, lp{W0,W1,W2,a0,a1,b0,b1},
//  us{W0,W1,a0,b0,b1}, wc{W0,W1,a0,b0,b1}
// ---------------------------------------------------------------------------
extern "C" void kernel_launch(void* const* d_in, const int* in_sizes, int n_in,
                              void* d_out, int out_size, void* d_ws, size_t ws_size,
                              hipStream_t stream)
{
    (void)in_sizes; (void)n_in; (void)out_size; (void)ws_size;

    const float* data   = (const float*)d_in[0];
    const int*   labels = (const int*)d_in[1];
    const float* bcW0f  = (const float*)d_in[2];
    const float* bcW1f  = (const float*)d_in[3];
    const float* bc_a0  = (const float*)d_in[4];
    const float* bc_b0  = (const float*)d_in[5];
    const float* bc_b1  = (const float*)d_in[6];
    const float* lpW0f  = (const float*)d_in[7];
    const float* lpW1f  = (const float*)d_in[8];
    const float* lpW2f  = (const float*)d_in[9];
    const float* lp_a0  = (const float*)d_in[10];
    const float* lp_a1  = (const float*)d_in[11];
    const float* lp_b0  = (const float*)d_in[12];
    const float* lp_b1  = (const float*)d_in[13];
    const float* usW0f  = (const float*)d_in[14];
    const float* usW1f  = (const float*)d_in[15];
    const float* us_a0  = (const float*)d_in[16];
    const float* us_b0  = (const float*)d_in[17];
    const float* us_b1  = (const float*)d_in[18];
    const float* wcW0f  = (const float*)d_in[19];
    const float* wcW1f  = (const float*)d_in[20];
    const float* wc_a0  = (const float*)d_in[21];
    const float* wc_b0  = (const float*)d_in[22];
    const float* wc_b1  = (const float*)d_in[23];

    char* ws = (char*)d_ws;
    size_t off = 0;
    auto alloc = [&](size_t bytes) -> void* {
        void* p = ws + off;
        off += (bytes + 255) & ~(size_t)255;
        return p;
    };
    float* wc_out = (float*)alloc((size_t)B_ * N_ * DW_ * 4);   // 16 MB
    float* usbuf  = (float*)alloc((size_t)B_ * N_ * DW_ * 4);   // 16 MB (us -> US in place)
    float* WCst   = (float*)alloc((size_t)B_ * KMAX_ * DW_ * 4);
    int*   cmbuf  = (int*)alloc((size_t)N_ * 4);
    __bf16* stepWp = (__bf16*)alloc((size_t)32768 * 2);  // packed step weights
    __bf16* wcW0p  = (__bf16*)alloc((size_t)C_ * DW_ * 2);
    __bf16* usW0p  = (__bf16*)alloc((size_t)C_ * DW_ * 2);
    __bf16* wcW1p  = (__bf16*)alloc((size_t)DW_ * DW_ * 2);
    __bf16* usW1p  = (__bf16*)alloc((size_t)DW_ * DW_ * 2);

    auto swz = [&](const float* s, __bf16* d, int K, int Nn) {
        int n = K * Nn;
        swizzle_b<<<(n + 255) / 256, 256, 0, stream>>>(s, d, K, Nn);
    };
    swz(wcW0f, wcW0p, C_, DW_);
    swz(usW0f, usW0p, C_, DW_);
    swz(wcW1f, wcW1p, DW_, DW_);
    swz(usW1f, usW1p, DW_, DW_);
    swz(bcW0f, stepWp + 0,     DW_, DBC_);
    swz(bcW1f, stepWp + 8192,  DBC_, DBC_);
    swz(lpW0f, stepWp + 24576, 192, 32);
    swz(lpW1f, stepWp + 30720, 32, 32);
    pack_w2_sw<<<2, 256, 0, stream>>>(lpW2f, stepWp + 31744);

    // front-end MLPs
    const size_t feLds = (size_t)(64 * 72 + 8192 + 64 * 136) * 2;
    frontend_mlp2<<<(B_ * N_) / 64, 256, feLds, stream>>>(
        data, wcW0p, wcW1p, usW0p, usW1p,
        wc_b0, wc_b1, wc_a0, us_b0, us_b1, us_a0, wc_out, usbuf);

    suffix_kernel<<<B_, DW_, 0, stream>>>(usbuf);
    cummax_kernel<<<1, 32, 0, stream>>>(labels, cmbuf);
    init_state<<<(B_ * KMAX_ * DW_ + 255) / 256, 256, 0, stream>>>(WCst, wc_out,
                                                                   (float*)d_out);

    // sequential scan: 2046 fused step kernels (captured into the graph once)
    const size_t STEP_BF16 = (size_t)32256 + 16000 + 19584;
    const size_t STEP_F32  = (size_t)144 * 132 + 128 + 64 + 80;
    const size_t stepLds = STEP_BF16 * 2 + STEP_F32 * 4;   // ~208 KB < 320 KB/WGP
    (void)hipFuncSetAttribute((const void*)step_kernel,
                              hipFuncAttributeMaxDynamicSharedMemorySize,
                              (int)stepLds);

    for (int t = 0; t < N_ - 2; ++t) {
        step_kernel<<<B_, 256, stepLds, stream>>>(
            t, wc_out, usbuf, labels, cmbuf, WCst, stepWp,
            bc_b0, bc_b1, bc_a0, lp_b0, lp_b1, lp_a0, lp_a1,
            (float*)d_out);
    }
}
